// Model_15247133901615
// MI455X (gfx1250) — matrix operand order repaired and checked by reading, not compiled, not run on hardware
//
#include <hip/hip_runtime.h>
#include <math.h>

#define BB 2048
#define TT 48
#define DD 128
#define HH 256
#define H4 (4*HH)
#define EPSL 1e-05f
#define IWF 0.3f

typedef __bf16 bf_t;
typedef __attribute__((ext_vector_type(16))) __bf16 v16bf;
typedef __attribute__((ext_vector_type(8)))  float  v8f;

union AFrag { v16bf v; unsigned short u[16]; };

__device__ __forceinline__ unsigned short f2bf(float f) {
    unsigned int u = __float_as_uint(f);
    u = (u + 0x7FFFu + ((u >> 16) & 1u)) >> 16;
    return (unsigned short)u;
}
__device__ __forceinline__ float sigf(float x) { return 1.0f / (1.0f + __expf(-x)); }

// ---------------------------------------------------------------------------
// WMMA GEMM:  C[M,N](f32) = A_bf16[M,K] @ W_bf16[N,K]^T (+bias[n]) (+= C if acc)
// A row m at A + aoff + m*lda (bf16, K contiguous). W packed (N,K).
// Block: 256 thr = 8 waves; tile 64(M) x 64(N); wave = 16(M) x 32(N) (2 WMMA
// tiles sharing one A fragment). All element offsets are multiples of 4 bf16
// -> 8B-aligned uint2 global loads / b64 LDS stores.
// M%64==0, N%64==0, K%32==0 (true for all uses here).
// ---------------------------------------------------------------------------
#define GTM 64
#define GTN 64
#define GTK 32

__global__ __launch_bounds__(256) void k_gemm(
    const unsigned short* __restrict__ A, long long aoff, int lda,
    const unsigned short* __restrict__ W,
    const float* __restrict__ bias,
    float* __restrict__ C, int ldc,
    int K, int accflag)
{
    __shared__ unsigned short As[GTM][GTK];
    __shared__ unsigned short Ws[GTN][GTK];
    const int tid  = threadIdx.x;
    const int lane = tid & 31;
    const int wave = tid >> 5;
    const int m0   = blockIdx.y * GTM;
    const int n0   = blockIdx.x * GTN;
    const int wm   = (wave & 3) * 16;
    const int wn   = (wave >> 2) * 32;
    const int half = lane >> 4;
    const int mr   = lane & 15;

    v8f acc0 = {};
    v8f acc1 = {};
    for (int k0 = 0; k0 < K; k0 += GTK) {
        // A tile: 64x32 = 512 uint2 chunks, 2 per thread
        #pragma unroll
        for (int i = 0; i < 2; ++i) {
            int idx = i * 256 + tid;
            int r = idx >> 3, cc = (idx & 7) * 4;
            *(uint2*)&As[r][cc] =
                *(const uint2*)(A + aoff + (long long)(m0 + r) * lda + (k0 + cc));
        }
        // W tile: 64x32 = 512 uint2 chunks, 2 per thread
        #pragma unroll
        for (int i = 0; i < 2; ++i) {
            int idx = i * 256 + tid;
            int r = idx >> 3, cc = (idx & 7) * 4;
            *(uint2*)&Ws[r][cc] =
                *(const uint2*)(W + (long long)(n0 + r) * K + (k0 + cc));
        }
        __syncthreads();
        AFrag af, bf0, bf1;
        #pragma unroll
        for (int e = 0; e < 16; ++e) {
            int ka = ((e < 8) ? e : e + 8) + half * 8;   // ISA A 16x32 layout
            af.u[e] = As[wm + mr][ka];
            int kb = ((e < 8) ? e : e + 8) + half * 8;   // ISA B 32x16 layout
            bf0.u[e] = Ws[wn + mr][kb];
            bf1.u[e] = Ws[wn + 16 + mr][kb];
        }
        acc0 = __builtin_amdgcn_wmma_f32_16x16x32_bf16(
            false, af.v, false, bf0.v, (short)0, acc0, false, false);
        acc1 = __builtin_amdgcn_wmma_f32_16x16x32_bf16(
            false, af.v, false, bf1.v, (short)0, acc1, false, false);
        __syncthreads();
    }
    const int n_a = n0 + wn + mr;
    const int n_b = n_a + 16;
    const float bv_a = bias ? bias[n_a] : 0.0f;
    const float bv_b = bias ? bias[n_b] : 0.0f;
    #pragma unroll
    for (int r = 0; r < 8; ++r) {
        int m = m0 + wm + r + 8 * half;                  // ISA C/D layout
        long long rowc = (long long)m * ldc;
        float va = acc0[r] + bv_a;
        float vb = acc1[r] + bv_b;
        if (accflag) { va += C[rowc + n_a]; vb += C[rowc + n_b]; }
        C[rowc + n_a] = va;
        C[rowc + n_b] = vb;
    }
}

// ---------------------------------------------------------------------------
// Elementwise / utility kernels
// ---------------------------------------------------------------------------
__global__ void k_zero(float* p, long long n) {
    long long i = (long long)blockIdx.x * blockDim.x + threadIdx.x;
    if (i < n) p[i] = 0.0f;
}
__global__ void k_f2bf(const float* __restrict__ x, unsigned short* __restrict__ y, long long n) {
    long long i = (long long)blockIdx.x * blockDim.x + threadIdx.x;
    if (i < n) y[i] = f2bf(x[i]);
}
__global__ void k_addvec(const float* a, const float* b, float* o, int n) {
    int i = blockIdx.x * blockDim.x + threadIdx.x;
    if (i < n) o[i] = a[i] + b[i];
}
// weight (N,K) slice from (N,ldw) at col koff, optional zero diagonal (n==k)
__global__ void k_wcvt(const float* __restrict__ W, unsigned short* __restrict__ o,
                       int N, int K, int ldw, int koff, int zdiag) {
    int i = blockIdx.x * blockDim.x + threadIdx.x;
    if (i >= N * K) return;
    int n = i / K, k = i % K;
    float v = W[(long long)n * ldw + koff + k];
    if (zdiag && n == k) v = 0.0f;
    o[i] = f2bf(v);
}
// feat[b,s,d] += sum_{t!=s} Wt[s,t]*v[b,t,d] + tb[s]     (block = b*T+s, thr = d)
__global__ __launch_bounds__(DD) void k_temporal(const float* __restrict__ v,
                                                 const float* __restrict__ Wt,
                                                 const float* __restrict__ tb,
                                                 float* __restrict__ feat) {
    __shared__ float w[TT];
    int b = blockIdx.x / TT, s = blockIdx.x % TT, d = threadIdx.x;
    if (d < TT) w[d] = (d == s) ? 0.0f : Wt[s * TT + d];
    __syncthreads();
    float sum = tb[s];
    const float* vb = v + (long long)b * TT * DD + d;
    #pragma unroll 4
    for (int t = 0; t < TT; ++t) sum += w[t] * vb[(long long)t * DD];
    feat[((long long)b * TT + s) * DD + d] += sum;
}
__global__ void k_vhat(const float* __restrict__ v, const float* __restrict__ m,
                       const float* __restrict__ feat, unsigned short* __restrict__ vh,
                       long long n) {
    long long i = (long long)blockIdx.x * blockDim.x + threadIdx.x;
    if (i < n) vh[i] = f2bf(v[i] * m[i] + feat[i] * (1.0f - m[i]));
}
// h *= exp(-|G|) ; h_bf = bf16(h)
__global__ void k_gamma(const float* __restrict__ G, float* __restrict__ h,
                        unsigned short* __restrict__ hbf, long long n) {
    long long i = (long long)blockIdx.x * blockDim.x + threadIdx.x;
    if (i < n) {
        float x = h[i] * __expf(-fabsf(G[i]));
        h[i] = x; hbf[i] = f2bf(x);
    }
}
__global__ void k_lstm(const float* __restrict__ g, float* __restrict__ h,
                       float* __restrict__ c, unsigned short* __restrict__ hbf) {
    long long i = (long long)blockIdx.x * blockDim.x + threadIdx.x;
    if (i >= (long long)BB * HH) return;
    int b = (int)(i / HH), j = (int)(i % HH);
    const float* gb = g + (long long)b * H4;
    float ig = sigf(gb[j]);
    float fg = sigf(gb[HH + j]);
    float gg = tanhf(gb[2 * HH + j]);
    float og = sigf(gb[3 * HH + j]);
    float cc = fg * c[i] + ig * gg;
    float hh = og * tanhf(cc);
    c[i] = cc; h[i] = hh; hbf[i] = f2bf(hh);
}
// per time step t of (B,T,D) mask: inv_sm[blockIdx] = 1/(sum+eps)
__global__ __launch_bounds__(256) void k_sminv(const float* __restrict__ m, float* __restrict__ inv) {
    __shared__ float red[256];
    int t = blockIdx.x, tid = threadIdx.x;
    float s = 0.0f;
    for (long long i = tid; i < (long long)BB * DD; i += 256) {
        long long b = i / DD, d = i % DD;
        s += m[(b * TT + t) * DD + d];
    }
    red[tid] = s; __syncthreads();
    for (int k = 128; k > 0; k >>= 1) { if (tid < k) red[tid] += red[tid + k]; __syncthreads(); }
    if (tid == 0) inv[t] = 1.0f / (red[0] + EPSL);
}
// decoder stage A: x_c from x_h; l1 accumulation
__global__ __launch_bounds__(256) void k_dec_a(
    const float* __restrict__ v, const float* __restrict__ m, long long toff,
    const float* __restrict__ xh, unsigned short* __restrict__ xcbf,
    const float* __restrict__ inv_t, float* __restrict__ accLoss) {
    __shared__ float red[256];
    int tid = threadIdx.x;
    long long i = (long long)blockIdx.x * 256 + tid;
    int b = (int)(i / DD), d = (int)(i % DD);
    long long si = (long long)b * TT * DD + toff + d;
    float x = v[si], mm = m[si], p = xh[i];
    xcbf[i] = f2bf(mm * x + (1.0f - mm) * p);
    red[tid] = fabsf(x - p) * mm; __syncthreads();
    for (int k = 128; k > 0; k >>= 1) { if (tid < k) red[tid] += red[tid + k]; __syncthreads(); }
    if (tid == 0) atomicAdd(accLoss, red[0] * (*inv_t));
}
// decoder stage B: alpha, c_h, c_c, l2/l3, imp write (mode0) or diff+avg (mode1)
__global__ __launch_bounds__(256) void k_dec_b(
    const float* __restrict__ v, const float* __restrict__ m, long long toff,
    const float* __restrict__ xh, const float* __restrict__ zh, const float* __restrict__ al,
    unsigned short* __restrict__ ccbf, float* __restrict__ imp, int timp, int mode,
    const float* __restrict__ inv_t, float* __restrict__ accLoss,
    float* __restrict__ ls_t, float* __restrict__ accDiff) {
    __shared__ float r0[256], r1[256], r2[256];
    int tid = threadIdx.x;
    long long i = (long long)blockIdx.x * 256 + tid;
    int b = (int)(i / DD), d = (int)(i % DD);
    long long si = (long long)b * TT * DD + toff + d;
    float x = v[si], mm = m[si];
    float px = xh[i], pz = zh[i];
    float a  = sigf(al[i]);
    float ch = a * pz + (1.0f - a) * px;
    float cc = mm * x + (1.0f - mm) * ch;
    ccbf[i] = f2bf(cc);
    long long io = ((long long)b * TT + timp) * DD + d;
    float dif = 0.0f;
    if (mode == 0) { imp[io] = cc; }
    else { float pb = imp[io]; dif = fabsf(cc - pb); imp[io] = 0.5f * (cc + pb); }
    r0[tid] = fabsf(x - pz) * mm;
    r1[tid] = fabsf(x - ch) * mm;
    r2[tid] = dif;
    __syncthreads();
    for (int k = 128; k > 0; k >>= 1) {
        if (tid < k) { r0[tid] += r0[tid + k]; r1[tid] += r1[tid + k]; r2[tid] += r2[tid + k]; }
        __syncthreads();
    }
    if (tid == 0) {
        float inv = *inv_t;
        atomicAdd(accLoss, (r0[0] + r1[0]) * inv);
        atomicAdd(ls_t, r1[0] * inv);
        if (mode) atomicAdd(accDiff, r2[0]);
    }
}
// output head: block per batch row, 256 thr = H
__global__ __launch_bounds__(HH) void k_outhead(const float* __restrict__ h,
                                                const float* __restrict__ W,
                                                const float* __restrict__ ob,
                                                const float* __restrict__ labels,
                                                float* __restrict__ pred,
                                                float* __restrict__ accBce) {
    __shared__ float red[HH];
    int b = blockIdx.x, j = threadIdx.x;
    red[j] = h[(long long)b * HH + j] * W[j]; __syncthreads();
    for (int k = HH / 2; k > 0; k >>= 1) { if (j < k) red[j] += red[j + k]; __syncthreads(); }
    if (j == 0) {
        float logit = red[0] + ob[0];
        pred[b] = sigf(logit);
        float y = labels[b];
        float mx = fmaxf(-logit, 0.0f);
        float bce = logit - logit * y + mx + __logf(__expf(-mx) + __expf(-logit - mx));
        atomicAdd(accBce, bce);
    }
}
__global__ void k_final(const float* __restrict__ lsf, const float* __restrict__ lsb,
                        const float* __restrict__ acc, float* __restrict__ out) {
    int t = threadIdx.x;
    if (t < TT) out[1 + t] = 0.5f * (lsf[t] + lsb[t]);
    if (t == 0) {
        float loss_c = 0.1f * acc[2] / ((float)BB * TT * DD);
        float xloss = acc[0] + acc[1] + loss_c;
        float yloss = acc[3] / (float)BB;
        out[0] = IWF * xloss + (1.0f - IWF) * yloss;
    }
}

// ---------------------------------------------------------------------------
extern "C" void kernel_launch(void* const* d_in, const int* in_sizes, int n_in,
                              void* d_out, int out_size, void* d_ws, size_t ws_size,
                              hipStream_t stream) {
    const float* fv = (const float*)d_in[0];
    const float* fm = (const float*)d_in[1];
    const float* fd = (const float*)d_in[2];
    const float* bv = (const float*)d_in[3];
    const float* bm = (const float*)d_in[4];
    const float* bd = (const float*)d_in[5];
    const float* labels = (const float*)d_in[6];
    const float* P[36];
    for (int i = 0; i < 36; ++i) P[i] = (const float*)d_in[7 + i];
    // P: 0 enc_feat_W 1 enc_feat_b 2 enc_temp_W 3 enc_temp_b 4 enc_td_W 5 enc_td_b
    //    6 enc_Wih 7 enc_bih 8 enc_Whh 9 enc_bhh | 10.. f block | 22.. b block | 34 out_W 35 out_b

    const long long NBTD = (long long)BB * TT * DD;
    char* wp = (char*)d_ws;
    auto alloc = [&](size_t bytes) -> void* {
        void* p = (void*)wp; wp += (bytes + 255) & ~(size_t)255; return p;
    };
    unsigned short* fv_bf  = (unsigned short*)alloc(NBTD * 2);
    unsigned short* fm_bf  = (unsigned short*)alloc(NBTD * 2);
    unsigned short* fd_bf  = (unsigned short*)alloc(NBTD * 2);
    unsigned short* bm_bf  = (unsigned short*)alloc(NBTD * 2);
    unsigned short* bd_bf  = (unsigned short*)alloc(NBTD * 2);
    unsigned short* vhat_bf= (unsigned short*)alloc(NBTD * 2);
    float* featbuf = (float*)alloc(NBTD * 4);
    float* h   = (float*)alloc((size_t)BB * HH * 4);
    float* c   = (float*)alloc((size_t)BB * HH * 4);
    unsigned short* h_bf = (unsigned short*)alloc((size_t)BB * HH * 2);
    float* gates = (float*)alloc((size_t)BB * H4 * 4);   // also gamma-pre (B,H)
    float* xh = (float*)alloc((size_t)BB * DD * 4);
    float* zh = (float*)alloc((size_t)BB * DD * 4);
    float* al = (float*)alloc((size_t)BB * DD * 4);
    unsigned short* xc_bf = (unsigned short*)alloc((size_t)BB * DD * 2);
    unsigned short* cc_bf = (unsigned short*)alloc((size_t)BB * DD * 2);
    // weights (bf16 packed (N,K))
    unsigned short* e_featW = (unsigned short*)alloc((size_t)DD * DD * 2);
    unsigned short* e_tdW   = (unsigned short*)alloc((size_t)HH * DD * 2);
    unsigned short* e_WihX  = (unsigned short*)alloc((size_t)H4 * DD * 2);
    unsigned short* e_WihM  = (unsigned short*)alloc((size_t)H4 * DD * 2);
    unsigned short* e_Whh   = (unsigned short*)alloc((size_t)H4 * HH * 2);
    float* e_bihh = (float*)alloc(H4 * 4);
    unsigned short *dtdW[2], *dhistW[2], *dfeatW[2], *dwWd[2], *dwWm[2], *dWihX[2], *dWihM[2], *dWhh[2];
    float* dbihh[2];
    for (int d = 0; d < 2; ++d) {
        dtdW[d]  = (unsigned short*)alloc((size_t)HH * DD * 2);
        dhistW[d]= (unsigned short*)alloc((size_t)DD * HH * 2);
        dfeatW[d]= (unsigned short*)alloc((size_t)DD * DD * 2);
        dwWd[d]  = (unsigned short*)alloc((size_t)DD * DD * 2);
        dwWm[d]  = (unsigned short*)alloc((size_t)DD * DD * 2);
        dWihX[d] = (unsigned short*)alloc((size_t)H4 * DD * 2);
        dWihM[d] = (unsigned short*)alloc((size_t)H4 * DD * 2);
        dWhh[d]  = (unsigned short*)alloc((size_t)H4 * HH * 2);
        dbihh[d] = (float*)alloc(H4 * 4);
    }
    float* acc   = (float*)alloc(16 * 4);   // 0 loss_f 1 loss_b 2 absdiff 3 bce
    float* lsf   = (float*)alloc(TT * 4);
    float* lsb   = (float*)alloc(TT * 4);
    float* inv_f = (float*)alloc(TT * 4);
    float* inv_b = (float*)alloc(TT * 4);

    float* out      = (float*)d_out;
    float* out_imp  = out + 1 + TT;
    float* out_pred = out + 1 + TT + NBTD;

    auto gemm = [&](const unsigned short* A, long long aoff, int lda,
                    const unsigned short* W, const float* bias, float* C,
                    int M, int N, int K, int accf) {
        dim3 g(N / GTN, M / GTM);
        k_gemm<<<g, 256, 0, stream>>>(A, aoff, lda, W, bias, C, N, K, accf);
    };
    auto zero = [&](float* p, long long n) {
        k_zero<<<(unsigned)((n + 255) / 256), 256, 0, stream>>>(p, n);
    };

    // ---- weight prep ----
    auto wc = [&](const float* W, unsigned short* o, int N, int K, int ldw, int koff, int zd) {
        k_wcvt<<<(N * K + 255) / 256, 256, 0, stream>>>(W, o, N, K, ldw, koff, zd);
    };
    wc(P[0], e_featW, DD, DD, DD, 0, 1);
    wc(P[4], e_tdW, HH, DD, DD, 0, 0);
    wc(P[6], e_WihX, H4, DD, 2 * DD, 0, 0);
    wc(P[6], e_WihM, H4, DD, 2 * DD, DD, 0);
    wc(P[8], e_Whh, H4, HH, HH, 0, 0);
    k_addvec<<<(H4 + 255) / 256, 256, 0, stream>>>(P[7], P[9], e_bihh, H4);
    for (int d = 0; d < 2; ++d) {
        int pb = 10 + 12 * d;                 // d=0 -> 'f', d=1 -> 'b'
        wc(P[pb + 0], dtdW[d], HH, DD, DD, 0, 0);
        wc(P[pb + 2], dhistW[d], DD, HH, HH, 0, 0);
        wc(P[pb + 4], dfeatW[d], DD, DD, DD, 0, 1);
        wc(P[pb + 6], dwWd[d], DD, DD, 2 * DD, 0, 0);
        wc(P[pb + 6], dwWm[d], DD, DD, 2 * DD, DD, 0);
        wc(P[pb + 8], dWihX[d], H4, DD, 2 * DD, 0, 0);
        wc(P[pb + 8], dWihM[d], H4, DD, 2 * DD, DD, 0);
        wc(P[pb + 10], dWhh[d], H4, HH, HH, 0, 0);
        k_addvec<<<(H4 + 255) / 256, 256, 0, stream>>>(P[pb + 9], P[pb + 11], dbihh[d], H4);
    }

    // ---- init state / accumulators / bf16 copies ----
    zero(h, (long long)BB * HH); zero(c, (long long)BB * HH);
    zero(acc, 16); zero(lsf, TT); zero(lsb, TT);
    unsigned nbtd_g = (unsigned)((NBTD + 255) / 256);
    k_f2bf<<<nbtd_g, 256, 0, stream>>>(fv, fv_bf, NBTD);
    k_f2bf<<<nbtd_g, 256, 0, stream>>>(fm, fm_bf, NBTD);
    k_f2bf<<<nbtd_g, 256, 0, stream>>>(fd, fd_bf, NBTD);
    k_f2bf<<<nbtd_g, 256, 0, stream>>>(bm, bm_bf, NBTD);
    k_f2bf<<<nbtd_g, 256, 0, stream>>>(bd, bd_bf, NBTD);
    k_sminv<<<TT, 256, 0, stream>>>(fm, inv_f);
    k_sminv<<<TT, 256, 0, stream>>>(bm, inv_b);

    // ---- encoder imputation: feat + temporal -> vhat ----
    gemm(fv_bf, 0, DD, e_featW, P[1], featbuf, BB * TT, DD, DD, 0);
    k_temporal<<<BB * TT, DD, 0, stream>>>(fv, P[2], P[3], featbuf);
    k_vhat<<<nbtd_g, 256, 0, stream>>>(fv, fm, featbuf, vhat_bf, NBTD);

    const int ldaT = TT * DD;
    const unsigned gBH = (unsigned)(((long long)BB * HH + 255) / 256);

    // ---- encoder LSTM over t = 0..T-2 ----
    for (int t = 0; t < TT - 1; ++t) {
        long long toff = (long long)t * DD;
        gemm(fd_bf, toff, ldaT, e_tdW, P[5], gates, BB, HH, DD, 0);
        k_gamma<<<gBH, 256, 0, stream>>>(gates, h, h_bf, (long long)BB * HH);
        gemm(vhat_bf, toff, ldaT, e_WihX, e_bihh, gates, BB, H4, DD, 0);
        gemm(fm_bf,   toff, ldaT, e_WihM, nullptr, gates, BB, H4, DD, 1);
        gemm(h_bf, 0, HH, e_Whh, nullptr, gates, BB, H4, HH, 1);
        k_lstm<<<gBH, 256, 0, stream>>>(gates, h, c, h_bf);
    }

    // ---- decoders: 'b' first (mode 0, time-reversed imp), then 'f' (mode 1) ----
    for (int pass = 0; pass < 2; ++pass) {
        int d      = pass ? 0 : 1;            // weight set: pass0 -> 'b'(1), pass1 -> 'f'(0)
        const float* vv = pass ? fv : bv;
        const float* mmv= pass ? fm : bm;
        const unsigned short* m_bf = pass ? fm_bf : bm_bf;
        const unsigned short* d_bf = pass ? fd_bf : bd_bf;
        const float* invs = pass ? inv_f : inv_b;
        float* ls   = pass ? lsf : lsb;
        float* accL = pass ? (acc + 0) : (acc + 1);
        int pb = 10 + 12 * d;
        unsigned gBD = (unsigned)(((long long)BB * DD + 255) / 256);
        for (int t = 0; t < TT; ++t) {
            long long toff = (long long)t * DD;
            gemm(d_bf, toff, ldaT, dtdW[d], P[pb + 1], gates, BB, HH, DD, 0);
            k_gamma<<<gBH, 256, 0, stream>>>(gates, h, h_bf, (long long)BB * HH);
            gemm(h_bf, 0, HH, dhistW[d], P[pb + 3], xh, BB, DD, HH, 0);
            k_dec_a<<<gBD, 256, 0, stream>>>(vv, mmv, toff, xh, xc_bf, invs + t, accL);
            gemm(xc_bf, 0, DD, dfeatW[d], P[pb + 5], zh, BB, DD, DD, 0);
            gemm(d_bf, toff, ldaT, dwWd[d], P[pb + 7], al, BB, DD, DD, 0);
            gemm(m_bf, toff, ldaT, dwWm[d], nullptr, al, BB, DD, DD, 1);
            int timp = pass ? t : (TT - 1 - t);
            k_dec_b<<<gBD, 256, 0, stream>>>(vv, mmv, toff, xh, zh, al, cc_bf,
                                             out_imp, timp, pass, invs + t, accL,
                                             ls + t, acc + 2);
            gemm(cc_bf, 0, DD, dWihX[d], dbihh[d], gates, BB, H4, DD, 0);
            gemm(m_bf, toff, ldaT, dWihM[d], nullptr, gates, BB, H4, DD, 1);
            gemm(h_bf, 0, HH, dWhh[d], nullptr, gates, BB, H4, HH, 1);
            k_lstm<<<gBH, 256, 0, stream>>>(gates, h, c, h_bf);
        }
    }

    // ---- output head + final assembly ----
    k_outhead<<<BB, HH, 0, stream>>>(h, P[34], P[35], labels, out_pred, acc + 3);
    k_final<<<1, 64, 0, stream>>>(lsf, lsb, acc, out);
}
